// SparseConvNeXtBlock_11845519802715
// MI455X (gfx1250) — compile-verified, matrix operand order checked by reading, not run on hardware
//

#include <hip/hip_runtime.h>
#include <hip/hip_bf16.h>
#include <stdint.h>

// ---------------------------------------------------------------------------
// SparseConvNeXt block for MI455X (gfx1250, wave32, WMMA bf16 16x16x32).
//
//   x:(16,384,56,56) f32  -> dwconv7x7 -> LN(C) -> [mask irrelevant pre-MLP]
//   -> GEMM1(384->1536)+GELU -> GEMM2(1536->384)*gamma -> mask -> x + y
//
// tokens M = 16*56*56 = 50176 = 392 * 128 (exact multiple of the M tile).
// GEMMs: 128x128x32 tiles, double-buffered LDS; A-tiles staged by the Tensor
// Data Mover (async, TENSORcnt) which also inserts the 16B/row LDS padding
// (pad_interval=3 -> every 8 qwords=64B, pad_amount=3 -> 4 dwords) producing
// 80B rows = LDS_STR bf16 elems. Tile k+1 is staged while tile k computes.
// ---------------------------------------------------------------------------

typedef __attribute__((ext_vector_type(16))) __bf16 v16bf;
typedef __attribute__((ext_vector_type(8)))  float  v8f;
typedef unsigned int u32x4 __attribute__((ext_vector_type(4)));
typedef int          i32x8 __attribute__((ext_vector_type(8)));
typedef int          i32x4 __attribute__((ext_vector_type(4)));

#define TOKENS   50176
#define CDIM     384
#define HID      1536
#define HW       3136     // 56*56
#define LDS_STR  40       // LDS row stride in bf16 elems: 80B rows
#define TILE_B   (128 * LDS_STR * 2)   // bytes per LDS tile buffer = 10240

// ---- workspace layout (bytes) ----
#define WS_W1B   0u                       // 384*1536*2  = 1179648
#define WS_W2B   1179648u                 // 1536*384*2  = 1179648
#define WS_A     2359296u                 // 50176*384*2 = 38535168
#define WS_H     40894464u                // 50176*1536*2 = 154140672

#define HAVE_TDM __has_builtin(__builtin_amdgcn_tensor_load_to_lds)

static __device__ __forceinline__ float gelu_exact(float v) {
    return 0.5f * v * (1.0f + erff(v * 0.70710678118654752f));
}

// Load one 16x32 bf16 WMMA A/B fragment from an LDS row (row-major, K contig).
// half = 0 for lanes 0-15 (K base 0 / 16), 1 for lanes 16-31 (K base 8 / 24).
// Per ISA 7.12.2: v0..v3 <- K = half*8 + 0..7 ; v4..v7 <- K = 16 + half*8 + 0..7
static __device__ __forceinline__ v16bf load_frag(const __bf16* row, int half) {
    union { v16bf v; uint2 q[4]; } u;
    const int kb = half << 3;
    u.q[0] = *(const uint2*)(row + kb);
    u.q[1] = *(const uint2*)(row + kb + 4);
    u.q[2] = *(const uint2*)(row + kb + 16);
    u.q[3] = *(const uint2*)(row + kb + 20);
    return u.v;
}

#if HAVE_TDM
// TDM: async copy of a 128-row x 32-bf16 tile (global row stride ldK elems)
// into LDS at byte offset ldsOff, with 16B padding appended per 64B row.
// D# per CDNA5 ISA sec 8.3/8.4 (group0: count/lds/global/type; group1: dims).
// This toolchain exposes the 6-arg builtin form:
//   (u32x4 g0, i32x8 g1, i32x4 g2, i32x4 g3, i32x8 extra, i32 cpol)
static __device__ __forceinline__ void tdm_load_tile(const __bf16* gsrc,
                                                     unsigned ldsOff, int ldK) {
    const unsigned long long ga = (unsigned long long)(size_t)gsrc;
    u32x4 g0;
    g0.x = 1u;                                          // count=1, user mode
    g0.y = ldsOff;                                      // lds_addr
    g0.z = (unsigned)(ga & 0xFFFFFFFFu);                // global_addr[31:0]
    g0.w = (unsigned)((ga >> 32) & 0x1FFFFFFu)          // global_addr[56:32]
         | 0x80000000u;                                 // type=2 (image)
    i32x8 g1;
    g1[0] = (1 << 16)     // data_size = 1 -> 2 bytes
          | (1 << 20)     // pad_enable
          | (3 << 22)     // pad_interval: 2^3 qwords = 64B
          | (3 << 25);    // pad_amount: 4 dwords = 16B
    g1[1] = (32  << 16);  // tensor_dim0 = 32  (bits 79:48, low half)
    g1[2] = (128 << 16);  // tensor_dim1 = 128 (bits 111:80, low half)
    g1[3] = (32  << 16);  // tile_dim0 = 32    (bits 127:112)
    g1[4] = 128;          // tile_dim1 = 128   (bits 143:128); tile_dim2 = 0
    g1[5] = ldK;          // tensor_dim0_stride[31:0] (bits 207:160)
    g1[6] = 0;
    g1[7] = 0;
    const i32x4 z4 = {0, 0, 0, 0};
    const i32x8 z8 = {0, 0, 0, 0, 0, 0, 0, 0};
    __builtin_amdgcn_tensor_load_to_lds(g0, g1, z4, z4, z8, 0);
}
#else
// Fallback: manual A-tile staging (128 x 32 bf16, LDS rows of LDS_STR elems).
static __device__ __forceinline__ void stage_a(const __bf16* A, int ldK,
                                               int m0, int kk, __bf16* as,
                                               int tid) {
#pragma unroll
    for (int it = 0; it < 2; ++it) {
        const int r = (tid >> 2) + it * 64;
        const int cs = (tid & 3) << 3;
        const uint4 d = *(const uint4*)(A + (size_t)(m0 + r) * ldK + kk + cs);
        *(uint4*)(&as[r * LDS_STR + cs]) = d;
    }
}
#endif

// B tile: 32 K x 128 N from row-major B (stride ldB), stored transposed
// into LDS -> bs[n][k] with row stride LDS_STR.
static __device__ __forceinline__ void stage_b(const __bf16* B, int ldB,
                                               int kk, int n0, __bf16* bs,
                                               int tid) {
#pragma unroll
    for (int it = 0; it < 2; ++it) {
        const int k  = (tid >> 4) + it * 16;
        const int n8 = (tid & 15) << 3;
        union { uint4 q; __bf16 e[8]; } u;
        u.q = *(const uint4*)(B + (size_t)(kk + k) * ldB + n0 + n8);
#pragma unroll
        for (int e = 0; e < 8; ++e) bs[(n8 + e) * LDS_STR + k] = u.e[e];
    }
}

// ---------------------------------------------------------------------------
// Kernel 0: fp32 -> bf16 weight conversion (w1 and w2 are both 589824 elems)
// ---------------------------------------------------------------------------
__global__ void convert_weights(const float* __restrict__ w1,
                                const float* __restrict__ w2,
                                __bf16* __restrict__ w1b,
                                __bf16* __restrict__ w2b, int n) {
    int i = blockIdx.x * blockDim.x + threadIdx.x;
    if (i < n) {
        w1b[i] = (__bf16)w1[i];
        w2b[i] = (__bf16)w2[i];
    }
}

// ---------------------------------------------------------------------------
// Kernel 1: depthwise 7x7 conv + bias + LayerNorm(C) -> bf16 A[token][c]
// One block per (b,h): 384 threads = one channel each, 12 waves.
// Sliding 7-wide register window: 7 loads/token (x is L2 resident).
// ---------------------------------------------------------------------------
__global__ __launch_bounds__(384) void conv_ln(
        const float* __restrict__ x,
        const float* __restrict__ dw_w, const float* __restrict__ dw_b,
        const float* __restrict__ ln_w, const float* __restrict__ ln_b,
        __bf16* __restrict__ A) {
    __shared__ float redS[12], redS2[12], bc[2];

    const int c    = threadIdx.x;          // channel, 0..383
    const int lane = c & 31;
    const int wid  = c >> 5;
    const int b    = blockIdx.x / 56;
    const int h    = blockIdx.x % 56;

    const float* xc = x + ((size_t)b * CDIM + c) * HW;
    const float dwb = dw_b[c];
    const float lnw = ln_w[c];
    const float lnb = ln_b[c];

    float wt[7][7];
    float win[7][7];
#pragma unroll
    for (int i = 0; i < 7; ++i) {
        const int hh = h + i - 3;
        const bool hok = (unsigned)hh < 56u;
#pragma unroll
        for (int j = 0; j < 7; ++j) wt[i][j] = dw_w[c * 49 + i * 7 + j];
#pragma unroll
        for (int j = 0; j < 6; ++j) {
            const int ww = j - 3;
            win[i][j] = (hok && ww >= 0) ? xc[hh * 56 + ww] : 0.0f;
        }
        win[i][6] = 0.0f;
    }

    for (int w = 0; w < 56; ++w) {
#pragma unroll
        for (int i = 0; i < 7; ++i) {
            const int hh = h + i - 3;
            const int ww = w + 3;
            win[i][6] = ((unsigned)hh < 56u && ww < 56) ? xc[hh * 56 + ww] : 0.0f;
        }
        float y = dwb;
#pragma unroll
        for (int i = 0; i < 7; ++i)
#pragma unroll
            for (int j = 0; j < 7; ++j) y = fmaf(win[i][j], wt[i][j], y);
#pragma unroll
        for (int i = 0; i < 7; ++i)
#pragma unroll
            for (int j = 0; j < 6; ++j) win[i][j] = win[i][j + 1];

        // LayerNorm over 384 channels: wave shuffle reduce + LDS combine
        float s = y, s2 = y * y;
#pragma unroll
        for (int off = 16; off >= 1; off >>= 1) {
            s  += __shfl_xor(s,  off, 32);
            s2 += __shfl_xor(s2, off, 32);
        }
        if (lane == 0) { redS[wid] = s; redS2[wid] = s2; }
        __syncthreads();
        if (threadIdx.x == 0) {
            float ts = 0.f, ts2 = 0.f;
#pragma unroll
            for (int i = 0; i < 12; ++i) { ts += redS[i]; ts2 += redS2[i]; }
            const float mu  = ts * (1.0f / 384.0f);
            const float var = ts2 * (1.0f / 384.0f) - mu * mu;
            bc[0] = mu;
            bc[1] = rsqrtf(var + 1e-6f);
        }
        __syncthreads();
        const float mu = bc[0], inv = bc[1];
        const float yn = (y - mu) * inv * lnw + lnb;
        const int tok = b * HW + h * 56 + w;
        A[(size_t)tok * CDIM + c] = (__bf16)yn;
    }
}

// ---------------------------------------------------------------------------
// Kernel 2: H = GELU(A @ w1 + b1), bf16 out.  M=50176 N=1536 K=384.
// Double-buffered: TDM + B-stage of tile k+1 overlap WMMA of tile k.
// ---------------------------------------------------------------------------
__global__ __launch_bounds__(256) void gemm1_gelu(
        const __bf16* __restrict__ A, const __bf16* __restrict__ B,
        const float* __restrict__ b1, __bf16* __restrict__ Hd) {
    __shared__ char smem[4 * TILE_B];          // As[2] | Bs[2]
    __bf16* AsB = (__bf16*)smem;
    __bf16* BsB = (__bf16*)(smem + 2 * TILE_B);

    const int tid  = threadIdx.x;
    const int lane = tid & 31;
    const int wid  = tid >> 5;
    const int wm   = wid & 1;
    const int wn   = wid >> 1;
    const int m0   = blockIdx.x * 128;
    const int n0   = blockIdx.y * 128;
    const int half = lane >> 4;
    const int lr   = lane & 15;
#if HAVE_TDM
    const unsigned asOff = (unsigned)(size_t)&AsB[0];  // LDS byte offset
#endif

    v8f acc[4][2] = {};

    // prologue: stage tile 0
#if HAVE_TDM
    if (wid == 0) tdm_load_tile(A + (size_t)m0 * CDIM, asOff, CDIM);
#else
    stage_a(A, CDIM, m0, 0, AsB, tid);
#endif
    stage_b(B, HID, 0, n0, BsB, tid);
#if HAVE_TDM
    if (wid == 0) __builtin_amdgcn_s_wait_tensorcnt(0);
#endif
    __syncthreads();

    int buf = 0;
    for (int kk = 0; kk < CDIM; kk += 32) {
        const int nbuf = buf ^ 1;
        const bool more = (kk + 32 < CDIM);
        if (more) {   // stage tile k+1 into the other buffer (overlaps WMMA)
#if HAVE_TDM
            if (wid == 0)
                tdm_load_tile(A + (size_t)m0 * CDIM + kk + 32,
                              asOff + (unsigned)(nbuf * TILE_B), CDIM);
#else
            stage_a(A, CDIM, m0, kk + 32, AsB + nbuf * (128 * LDS_STR), tid);
#endif
            stage_b(B, HID, kk + 32, n0, BsB + nbuf * (128 * LDS_STR), tid);
        }

        const __bf16* as = AsB + buf * (128 * LDS_STR);
        const __bf16* bs = BsB + buf * (128 * LDS_STR);
        v16bf af[4], bfr[2];
#pragma unroll
        for (int mt = 0; mt < 4; ++mt)
            af[mt] = load_frag(&as[(wm * 64 + mt * 16 + lr) * LDS_STR], half);
#pragma unroll
        for (int nt = 0; nt < 2; ++nt)
            bfr[nt] = load_frag(&bs[(wn * 32 + nt * 16 + lr) * LDS_STR], half);
#pragma unroll
        for (int mt = 0; mt < 4; ++mt)
#pragma unroll
            for (int nt = 0; nt < 2; ++nt)
                acc[mt][nt] = __builtin_amdgcn_wmma_f32_16x16x32_bf16(
                    false, af[mt], false, bfr[nt], (short)0, acc[mt][nt], false, false);

#if HAVE_TDM
        if (more && wid == 0) __builtin_amdgcn_s_wait_tensorcnt(0);
#endif
        __syncthreads();   // publish tile k+1; all reads of tile k are done
        buf = nbuf;
    }

    // epilogue: +bias, exact GELU, bf16 store
#pragma unroll
    for (int mt = 0; mt < 4; ++mt)
#pragma unroll
        for (int nt = 0; nt < 2; ++nt) {
            const int n = n0 + wn * 32 + nt * 16 + lr;
            const float bias = b1[n];
#pragma unroll
            for (int i = 0; i < 8; ++i) {
                const int m = m0 + wm * 64 + mt * 16 + ((lane < 16) ? i : i + 8);
                float v = acc[mt][nt][i] + bias;
                Hd[(size_t)m * HID + n] = (__bf16)gelu_exact(v);
            }
        }
}

// ---------------------------------------------------------------------------
// Kernel 3: out = x + mask * (gamma * (H @ w2 + b2)), NCHW f32.
// M=50176 N=384 K=1536. Double-buffered like gemm1; epilogue transposes
// 128x32-token chunks through LDS so NCHW stores / x loads coalesce along w.
// ---------------------------------------------------------------------------
__global__ __launch_bounds__(256) void gemm2_out(
        const __bf16* __restrict__ Hd, const __bf16* __restrict__ B,
        const float* __restrict__ b2, const float* __restrict__ gamma,
        const float* __restrict__ x, const unsigned char* __restrict__ mask,
        float* __restrict__ out) {
    __shared__ char smem[4 * TILE_B];          // As[2] | Bs[2]; T aliases
    __bf16* AsB = (__bf16*)smem;
    __bf16* BsB = (__bf16*)(smem + 2 * TILE_B);
    float*  T   = (float*)smem;                // 128 ch x 33 f32 = 16896 B

    const int tid  = threadIdx.x;
    const int lane = tid & 31;
    const int wid  = tid >> 5;
    const int wm   = wid & 1;
    const int wn   = wid >> 1;
    const int m0   = blockIdx.x * 128;
    const int n0   = blockIdx.y * 128;
    const int half = lane >> 4;
    const int lr   = lane & 15;
#if HAVE_TDM
    const unsigned asOff = (unsigned)(size_t)&AsB[0];
#endif

    v8f acc[4][2] = {};

#if HAVE_TDM
    if (wid == 0) tdm_load_tile(Hd + (size_t)m0 * HID, asOff, HID);
#else
    stage_a(Hd, HID, m0, 0, AsB, tid);
#endif
    stage_b(B, CDIM, 0, n0, BsB, tid);
#if HAVE_TDM
    if (wid == 0) __builtin_amdgcn_s_wait_tensorcnt(0);
#endif
    __syncthreads();

    int buf = 0;
    for (int kk = 0; kk < HID; kk += 32) {
        const int nbuf = buf ^ 1;
        const bool more = (kk + 32 < HID);
        if (more) {
#if HAVE_TDM
            if (wid == 0)
                tdm_load_tile(Hd + (size_t)m0 * HID + kk + 32,
                              asOff + (unsigned)(nbuf * TILE_B), HID);
#else
            stage_a(Hd, HID, m0, kk + 32, AsB + nbuf * (128 * LDS_STR), tid);
#endif
            stage_b(B, CDIM, kk + 32, n0, BsB + nbuf * (128 * LDS_STR), tid);
        }

        const __bf16* as = AsB + buf * (128 * LDS_STR);
        const __bf16* bs = BsB + buf * (128 * LDS_STR);
        v16bf af[4], bfr[2];
#pragma unroll
        for (int mt = 0; mt < 4; ++mt)
            af[mt] = load_frag(&as[(wm * 64 + mt * 16 + lr) * LDS_STR], half);
#pragma unroll
        for (int nt = 0; nt < 2; ++nt)
            bfr[nt] = load_frag(&bs[(wn * 32 + nt * 16 + lr) * LDS_STR], half);
#pragma unroll
        for (int mt = 0; mt < 4; ++mt)
#pragma unroll
            for (int nt = 0; nt < 2; ++nt)
                acc[mt][nt] = __builtin_amdgcn_wmma_f32_16x16x32_bf16(
                    false, af[mt], false, bfr[nt], (short)0, acc[mt][nt], false, false);

#if HAVE_TDM
        if (more && wid == 0) __builtin_amdgcn_s_wait_tensorcnt(0);
#endif
        __syncthreads();
        buf = nbuf;
    }

    // epilogue: 4 chunks of 32 tokens; LDS transpose to channel-major
    for (int c4 = 0; c4 < 4; ++c4) {
        __syncthreads();
#pragma unroll
        for (int mt = 0; mt < 4; ++mt) {
            const int rbase = wm * 64 + mt * 16;
            if (rbase >= c4 * 32 && rbase < c4 * 32 + 32) {
#pragma unroll
                for (int nt = 0; nt < 2; ++nt) {
                    const int ch = wn * 32 + nt * 16 + lr;
                    const int n  = n0 + ch;
                    const float bias = b2[n];
                    const float g    = gamma[n];
#pragma unroll
                    for (int i = 0; i < 8; ++i) {
                        const int r = rbase + ((lane < 16) ? i : i + 8) - c4 * 32;
                        T[ch * 33 + r] = (acc[mt][nt][i] + bias) * g;
                    }
                }
            }
        }
        __syncthreads();

        const int ch = tid >> 1;
        const int t0 = (tid & 1) << 4;
        const int c  = n0 + ch;
#pragma unroll
        for (int t = 0; t < 16; ++t) {
            const int tok = m0 + c4 * 32 + t0 + t;
            const int b   = tok / HW;
            const int rem = tok - b * HW;
            const int h   = rem / 56;
            const int w   = rem - h * 56;
            const unsigned char mv = mask[b * 49 + (h >> 3) * 7 + (w >> 3)];
            const float v = mv ? T[ch * 33 + t0 + t] : 0.0f;
            const size_t idx = (size_t)(b * CDIM + c) * HW + rem;
            out[idx] = x[idx] + v;
        }
    }
}

// ---------------------------------------------------------------------------
extern "C" void kernel_launch(void* const* d_in, const int* in_sizes, int n_in,
                              void* d_out, int out_size, void* d_ws, size_t ws_size,
                              hipStream_t stream) {
    const float* x    = (const float*)d_in[0];
    const unsigned char* mask = (const unsigned char*)d_in[1]; // jnp bool -> 1 byte
    const float* dw_w = (const float*)d_in[2];
    const float* dw_b = (const float*)d_in[3];
    const float* ln_w = (const float*)d_in[4];
    const float* ln_b = (const float*)d_in[5];
    const float* w1   = (const float*)d_in[6];
    const float* b1   = (const float*)d_in[7];
    const float* w2   = (const float*)d_in[8];
    const float* b2   = (const float*)d_in[9];
    const float* gam  = (const float*)d_in[10];
    float* out = (float*)d_out;

    char* ws = (char*)d_ws;
    __bf16* w1b = (__bf16*)(ws + WS_W1B);
    __bf16* w2b = (__bf16*)(ws + WS_W2B);
    __bf16* A   = (__bf16*)(ws + WS_A);
    __bf16* Hd  = (__bf16*)(ws + WS_H);

    const int nw = CDIM * HID; // 589824
    convert_weights<<<(nw + 255) / 256, 256, 0, stream>>>(w1, w2, w1b, w2b, nw);

    conv_ln<<<16 * 56, 384, 0, stream>>>(x, dw_w, dw_b, ln_w, ln_b, A);

    gemm1_gelu<<<dim3(TOKENS / 128, HID / 128), 256, 0, stream>>>(A, w1b, b1, Hd);

    gemm2_out<<<dim3(TOKENS / 128, CDIM / 128), 256, 0, stream>>>(
        Hd, w2b, b2, gam, x, mask, out);
}
